// ChannelSA3dCausal_83124797047423
// MI455X (gfx1250) — matrix operand order repaired and checked by reading, not compiled, not run on hardware
//
#include <hip/hip_runtime.h>

// ChannelSA3dCausal — CDNA5 (gfx1250) WMMA implementation.
// Phase 1: causal framed projection (GEMM K=32) -> q,k  (f16, layout [b][t][c][s])
// Phase 2: per-(b,t) channel attention Q(256x32) @ K^T -> softmax -> matvec -> residual.
// Both phases use V_WMMA_F32_16X16X32_F16 (f32 accumulate).

typedef __attribute__((ext_vector_type(16))) _Float16 v16h;
typedef __attribute__((ext_vector_type(8)))  float    v8f;

#define Bn 4
#define Cn 256
#define Tn 1024
#define Dn 32
#define Sn 32
#define SC_STRIDE 33   // 32 cols + 1 pad; gcd(33,64)=1 -> conflict-free row reads

// ---------------------------------------------------------------------------
// Kernel 1: projection. Grid: B*C*8 blocks of 256 threads (8 waves).
// Each wave handles one 16-row t-tile of one (b,c): 4 WMMAs (key/query x 2 s-tiles).
// ---------------------------------------------------------------------------
__global__ __launch_bounds__(256) void proj_kernel(
    const float* __restrict__ x,       // [B,C,T]
    const float* __restrict__ key_w,   // [S,D]
    const float* __restrict__ key_b,   // [S]
    const float* __restrict__ key_g,   // [S]
    const float* __restrict__ key_be,  // [S]
    const float* __restrict__ qry_w,
    const float* __restrict__ qry_b,
    const float* __restrict__ qry_g,
    const float* __restrict__ qry_be,
    _Float16* __restrict__ k_ws,       // [B,T,C,S] f16
    _Float16* __restrict__ q_ws)       // [B,T,C,S] f16
{
  const int lane = threadIdx.x & 31;
  const int wave = threadIdx.x >> 5;     // 0..7
  const int hg   = lane >> 4;            // half-group (lanes 16-31)
  const int lm   = lane & 15;

  int blk = blockIdx.x;                  // b*C*8 + c*8 + ttg
  const int ttg = blk & 7;  blk >>= 3;
  const int c   = blk % Cn;
  const int b   = blk / Cn;
  const int t0  = (ttg * 8 + wave) * 16; // 64 t-tiles total per (b,c)

  // ---- A fragment: framed[t, d] = xs[b,c, t + d - 31] (0 if OOB) ----
  // A k order used here (same as B): lane row M = lane%16; element i -> K = i + 16*hg
  const float* xrow = x + ((size_t)b * Cn + c) * Tn;
  const int trow = t0 + lm;
  v16h a;
#pragma unroll
  for (int i = 0; i < 16; ++i) {
    const int d  = i + (hg << 4);
    const int ts = trow + d - 31;
    const float v = (ts >= 0) ? xrow[ts] : 0.0f;   // ts <= T-1 always
    a[i] = (_Float16)v;
  }

  const float* wmat[2] = { key_w, qry_w };
  v8f accs[4];
#pragma unroll
  for (int p = 0; p < 2; ++p) {
#pragma unroll
    for (int st = 0; st < 2; ++st) {
      // B fragment: B[K=d][N=s] = w[s][d]; lane col N = st*16+lm; element i -> K = i + 16*hg
      const int s = st * 16 + lm;
      const float* wrow = wmat[p] + s * Dn + hg * 16;   // 16 contiguous floats
      v16h bm;
#pragma unroll
      for (int i = 0; i < 16; ++i) bm[i] = (_Float16)wrow[i];
      v8f acc = {};
      acc = __builtin_amdgcn_wmma_f32_16x16x32_f16(
                false, a, false, bm, (short)0, acc, false, false);
      accs[p * 2 + st] = acc;
    }
  }

  // ---- epilogue: h = relu((acc + bias)*g/sqrt(1+eps) + beta), store f16 ----
  const float inv = 1.0f / sqrtf(1.0f + 1e-5f);   // constant-folded
#pragma unroll
  for (int p = 0; p < 2; ++p) {
    const float* bias = p ? qry_b  : key_b;
    const float* gam  = p ? qry_g  : key_g;
    const float* bet  = p ? qry_be : key_be;
    _Float16*    dst  = p ? q_ws   : k_ws;
#pragma unroll
    for (int st = 0; st < 2; ++st) {
      const int   s  = st * 16 + lm;                 // C/D layout: N = lane%16
      const float gs = gam[s] * inv;
      const float bb = bias[s] * gs + bet[s];
      const v8f acc = accs[p * 2 + st];
#pragma unroll
      for (int r = 0; r < 8; ++r) {
        const int t = t0 + r + hg * 8;               // C/D layout: M = r + 8*hg
        float h = acc[r] * gs + bb;
        h = h > 0.0f ? h : 0.0f;
        dst[(((size_t)b * Tn + t) * Cn + c) * Sn + s] = (_Float16)h;
      }
    }
  }
}

// ---------------------------------------------------------------------------
// Kernel 2: per-(b,t) attention. Grid: B*T blocks of 256 threads (8 waves).
// 256x256 scores as 16x16 WMMA tiles; chunked (32-col) online softmax + matvec.
// ---------------------------------------------------------------------------
__global__ __launch_bounds__(256) void attn_kernel(
    const float*    __restrict__ x,     // [B,C,T]
    const _Float16* __restrict__ k_ws,  // [B,T,C,S]
    const _Float16* __restrict__ q_ws,  // [B,T,C,S]
    float*          __restrict__ out)   // [B,C,T]
{
  __shared__ float sc[Cn * SC_STRIDE];  // 256 x 32(+1 pad) score chunk (~34 KB)
  __shared__ float xcol[Cn];            // xs[b, :, t]

  const int tid  = threadIdx.x;
  const int lane = tid & 31;
  const int wave = tid >> 5;            // 0..7
  const int hg   = lane >> 4;
  const int lm   = lane & 15;

  const int bt = blockIdx.x;
  const int t  = bt % Tn;
  const int b  = bt / Tn;

  xcol[tid] = x[((size_t)b * Cn + tid) * Tn + t];

  const _Float16* qbase = q_ws + ((size_t)b * Tn + t) * (size_t)(Cn * Sn);
  const _Float16* kbase = k_ws + ((size_t)b * Tn + t) * (size_t)(Cn * Sn);
  const float scale = 0.17677669529663687f;   // 1/sqrt(32)

  // Preload this wave's two A fragments (Q row-blocks), reused across all j.
  v16h afr[2];
#pragma unroll
  for (int tr = 0; tr < 2; ++tr) {
    const int irow = (wave * 2 + tr) * 16 + lm;
    const _Float16* qp = qbase + (size_t)irow * Sn;
#pragma unroll
    for (int i = 0; i < 16; ++i) {
      const int s = i + (hg << 4);
      afr[tr][i] = qp[s];
    }
  }

  // Online softmax state: thread `tid` owns score row i = tid.
  float m_run = -3.402823466e38f;
  float l_run = 0.0f;
  float a_run = 0.0f;

  __syncthreads();   // xcol visible

  for (int jc = 0; jc < 8; ++jc) {          // 8 chunks of 32 columns
    const int jbase = jc * 32;

    // -- compute phase: each wave fills its 2 tile-rows x 2 j-tiles --
#pragma unroll
    for (int tr = 0; tr < 2; ++tr) {
      const int i0 = (wave * 2 + tr) * 16;
#pragma unroll
      for (int jl = 0; jl < 2; ++jl) {
        const int j0 = jbase + jl * 16;
        // B fragment: B[K=s][N=j] = k[j][s]; 16 contiguous f16 per lane
        const _Float16* kp = kbase + (size_t)(j0 + lm) * Sn + hg * 16;
        v16h bm;
#pragma unroll
        for (int i = 0; i < 16; ++i) bm[i] = kp[i];
        v8f acc = {};
        acc = __builtin_amdgcn_wmma_f32_16x16x32_f16(
                  false, afr[tr], false, bm, (short)0, acc, false, false);
#pragma unroll
        for (int r = 0; r < 8; ++r)
          sc[(i0 + r + hg * 8) * SC_STRIDE + (jl * 16 + lm)] = acc[r] * scale;
      }
    }
    __syncthreads();

    // -- update phase: per-row online softmax over this 32-col chunk --
    {
      const float* row = &sc[tid * SC_STRIDE];
      float cm = -3.402823466e38f;
#pragma unroll
      for (int jj = 0; jj < 32; ++jj) cm = fmaxf(cm, row[jj]);
      const float nm = fmaxf(m_run, cm);
      const float f  = __expf(m_run - nm);
      l_run *= f;
      a_run *= f;
#pragma unroll
      for (int jj = 0; jj < 32; ++jj) {
        const float e = __expf(row[jj] - nm);
        l_run += e;
        a_run += e * xcol[jbase + jj];
      }
      m_run = nm;
    }
    __syncthreads();   // before chunk buffer is overwritten
  }

  const size_t oi = ((size_t)b * Cn + tid) * Tn + t;
  out[oi] = x[oi] + a_run / l_run;
}

// ---------------------------------------------------------------------------
extern "C" void kernel_launch(void* const* d_in, const int* in_sizes, int n_in,
                              void* d_out, int out_size, void* d_ws, size_t ws_size,
                              hipStream_t stream) {
  const float* x      = (const float*)d_in[0];
  const float* key_w  = (const float*)d_in[1];
  const float* key_b  = (const float*)d_in[2];
  const float* key_g  = (const float*)d_in[3];
  const float* key_be = (const float*)d_in[4];
  const float* qry_w  = (const float*)d_in[5];
  const float* qry_b  = (const float*)d_in[6];
  const float* qry_g  = (const float*)d_in[7];
  const float* qry_be = (const float*)d_in[8];
  float* out = (float*)d_out;

  // Workspace: q/k in f16, [B,T,C,S] each = 64 MiB (128 MiB total).
  const size_t elems = (size_t)Bn * Tn * Cn * Sn;
  _Float16* q_ws = (_Float16*)d_ws;
  _Float16* k_ws = q_ws + elems;

  proj_kernel<<<Bn * Cn * 8, 256, 0, stream>>>(
      x, key_w, key_b, key_g, key_be, qry_w, qry_b, qry_g, qry_be, k_ws, q_ws);

  attn_kernel<<<Bn * Tn, 256, 0, stream>>>(x, k_ws, q_ws, out);
}
